// Attention_78752520339750
// MI455X (gfx1250) — matrix operand order repaired and checked by reading, not compiled, not run on hardware
//
#include <hip/hip_runtime.h>
#include <hip/hip_bf16.h>

// Problem constants (from reference): N=4096 tokens, H=32, HKV=8, D=128, S=2048
#define NTOK 4096
#define NH   32
#define NKV  8
#define HD   128
#define SEQ  2048
#define GQA  (NH / NKV)          // 4
#define QTILE 16
#define KTILE 32
#define ATTN_SCALE 0.08838834764831845f   // 1/sqrt(128)
#define NEG_BIG (-1e30f)

typedef __attribute__((ext_vector_type(16))) __bf16 v16bf;
typedef __attribute__((ext_vector_type(8)))  __bf16 v8bf;
typedef __attribute__((ext_vector_type(8)))  float  v8f;

// LDS row strides (bf16 elements): multiples of 16 elems (32B) for aligned
// ds_load_b128 pairs, non-power-of-two to spread banks.
#define KSTR 144   // K tile:  32 rows x 128 cols
#define VSTR 48    // V tile (transposed): 128 rows x 32 cols
#define PSTR 48    // P scratch per wave: 16 rows x 32 cols

__global__ void kv_scatter_kernel(const float* __restrict__ k,
                                  const float* __restrict__ v,
                                  const int* __restrict__ slot,
                                  float* __restrict__ kc,
                                  float* __restrict__ vc,
                                  int total) {
    int i = blockIdx.x * blockDim.x + threadIdx.x;
    if (i >= total) return;
    int t = i >> 10;          // / (NKV*HD) == 1024
    int c = i & 1023;
    int s = slot[t];
    kc[(size_t)s * 1024 + c] = k[i];
    vc[(size_t)s * 1024 + c] = v[i];
}

// ---- cooperative K/V tile load: K row-major bf16, V transposed bf16 ----
__device__ __forceinline__ void load_tile(const float* __restrict__ k,
                                          const float* __restrict__ v,
                                          __bf16* Kt, __bf16* Vt,
                                          int b, int kvh, int kbase, int tid) {
    const int row = tid >> 2;             // 0..31 (key within tile)
    const int d0  = (tid & 3) * 32;       // 0/32/64/96
    const int key = kbase + row;
    const bool valid = key < SEQ;
    const float* kp = k + (((size_t)(b * SEQ + key)) * NKV + kvh) * HD + d0;
    const float* vp = v + (((size_t)(b * SEQ + key)) * NKV + kvh) * HD + d0;
    const float4 z4 = {0.f, 0.f, 0.f, 0.f};
    #pragma unroll
    for (int i = 0; i < 8; ++i) {
        const float4 fk = valid ? *(const float4*)(kp + 4 * i) : z4;
        const float4 fv = valid ? *(const float4*)(vp + 4 * i) : z4;
        __bf16* kd = &Kt[row * KSTR + d0 + 4 * i];
        kd[0] = (__bf16)fk.x; kd[1] = (__bf16)fk.y;
        kd[2] = (__bf16)fk.z; kd[3] = (__bf16)fk.w;
        const int dd = d0 + 4 * i;
        Vt[(dd + 0) * VSTR + row] = (__bf16)fv.x;
        Vt[(dd + 1) * VSTR + row] = (__bf16)fv.y;
        Vt[(dd + 2) * VSTR + row] = (__bf16)fv.z;
        Vt[(dd + 3) * VSTR + row] = (__bf16)fv.w;
    }
}

// ---- S = Q K^T : two 16x16 fp32 score tiles, software-pipelined B loads ----
__device__ __forceinline__ void qk_block(const __bf16* Kt, const v16bf (&qa)[4],
                                         v8f& s0, v8f& s1, int nl, int hs) {
    const v8f vzero = {0.f, 0.f, 0.f, 0.f, 0.f, 0.f, 0.f, 0.f};
    s0 = vzero; s1 = vzero;
    v16bf b0 = *(const v16bf*)&Kt[nl * KSTR + 16 * hs];
    v16bf b1 = *(const v16bf*)&Kt[(16 + nl) * KSTR + 16 * hs];
    #pragma unroll
    for (int c = 0; c < 4; ++c) {
        v16bf n0 = b0, n1 = b1;
        if (c < 3) {
            n0 = *(const v16bf*)&Kt[nl * KSTR + 32 * (c + 1) + 16 * hs];
            n1 = *(const v16bf*)&Kt[(16 + nl) * KSTR + 32 * (c + 1) + 16 * hs];
        }
        s0 = __builtin_amdgcn_wmma_f32_16x16x32_bf16(false, qa[c], false, b0,
                                                     (short)0, s0, false, false);
        s1 = __builtin_amdgcn_wmma_f32_16x16x32_bf16(false, qa[c], false, b1,
                                                     (short)0, s1, false, false);
        b0 = n0; b1 = n1;
    }
}

// ---- online softmax + O += P V (MASK only on the diagonal block) ----
template <bool MASK>
__device__ __forceinline__ void softmax_pv_block(v8f& s0, v8f& s1,
                                                 const __bf16* Vt, __bf16* Ptw,
                                                 v8f (&acc)[8],
                                                 float (&rm)[8], float (&rl)[8],
                                                 int kbase, int qbase, int nl, int hs) {
    const int key0 = kbase + nl;
    const int key1 = kbase + 16 + nl;
    const int row0 = qbase + 8 * hs;
    #pragma unroll
    for (int r = 0; r < 8; ++r) {
        float a0 = s0[r];
        float a1 = s1[r];
        if (MASK) {
            const int rowg = row0 + r;
            if (key0 > rowg) a0 = NEG_BIG;
            if (key1 > rowg) a1 = NEG_BIG;
        }
        float mx = fmaxf(a0, a1);
        mx = fmaxf(mx, __shfl_xor(mx, 1, 32));
        mx = fmaxf(mx, __shfl_xor(mx, 2, 32));
        mx = fmaxf(mx, __shfl_xor(mx, 4, 32));
        mx = fmaxf(mx, __shfl_xor(mx, 8, 32));
        const float mn = fmaxf(rm[r], mx);
        const float alpha = __expf(rm[r] - mn);
        rm[r] = mn;
        const float p0 = __expf(a0 - mn);
        const float p1 = __expf(a1 - mn);
        float ps = p0 + p1;
        ps += __shfl_xor(ps, 1, 32);
        ps += __shfl_xor(ps, 2, 32);
        ps += __shfl_xor(ps, 4, 32);
        ps += __shfl_xor(ps, 8, 32);
        rl[r] = rl[r] * alpha + ps;
        #pragma unroll
        for (int j = 0; j < 8; ++j) acc[j][r] *= alpha;   // rescale row r
        Ptw[(r + 8 * hs) * PSTR + nl]      = (__bf16)p0;
        Ptw[(r + 8 * hs) * PSTR + 16 + nl] = (__bf16)p1;
    }

    // P (C-layout) -> A-layout via per-wave LDS scratch (in-order within wave)
    const v8bf plo = *(const v8bf*)&Ptw[nl * PSTR + 16 * hs];
    const v8bf phi = *(const v8bf*)&Ptw[nl * PSTR + 16 * hs + 8];
    const v16bf pa = __builtin_shufflevector(plo, phi,
        0, 1, 2, 3, 4, 5, 6, 7, 8, 9, 10, 11, 12, 13, 14, 15);
    v16bf vb = *(const v16bf*)&Vt[nl * VSTR + 16 * hs];
    #pragma unroll
    for (int j = 0; j < 8; ++j) {
        v16bf nv = vb;
        if (j < 7) nv = *(const v16bf*)&Vt[(16 * (j + 1) + nl) * VSTR + 16 * hs];
        acc[j] = __builtin_amdgcn_wmma_f32_16x16x32_bf16(false, pa, false, vb,
                                                         (short)0, acc[j], false, false);
        vb = nv;
    }
}

__global__ __launch_bounds__(128) void fa_fwd_kernel(const float* __restrict__ q,
                                                     const float* __restrict__ k,
                                                     const float* __restrict__ v,
                                                     float* __restrict__ out) {
    // Double-buffered K/V tiles (ping-pong) + per-wave P scratch: ~48KB LDS.
    __shared__ __attribute__((aligned(32))) __bf16 Kt[2][KTILE * KSTR];
    __shared__ __attribute__((aligned(32))) __bf16 Vt[2][HD * VSTR];
    __shared__ __attribute__((aligned(32))) __bf16 Pt[4][QTILE * PSTR];

    const int nqt   = SEQ / QTILE;                // 128
    const int qtile = blockIdx.x % nqt;
    const int kvh   = (blockIdx.x / nqt) % NKV;
    const int b     = blockIdx.x / (nqt * NKV);

    const int tid  = threadIdx.x;
    const int wave = tid >> 5;
    const int lane = tid & 31;
    const int nl   = lane & 15;                   // column / row-in-half index
    const int hs   = lane >> 4;                   // which 16-lane half
    const int head = kvh * GQA + wave;            // this wave's q head
    const int qbase = qtile * QTILE;

    // ---- Q tile -> registers in WMMA A-layout, bf16, softmax scale folded in
    v16bf qa[4];
    {
        const float* qrow = q + (((size_t)(b * SEQ + qbase + nl)) * NH + head) * HD;
        #pragma unroll
        for (int c = 0; c < 4; ++c) {
            const int d0 = 32 * c + 16 * hs;      // lane<16: K=0..7 ; lane>=16: K=16..23
            const float4 f0 = *(const float4*)(qrow + d0);
            const float4 f1 = *(const float4*)(qrow + d0 + 4);
            const float4 f2 = *(const float4*)(qrow + d0 + 8);  // K=+8..15 / +24..31
            const float4 f3 = *(const float4*)(qrow + d0 + 12);
            v16bf a;
            a[0]=(__bf16)(f0.x*ATTN_SCALE);  a[1]=(__bf16)(f0.y*ATTN_SCALE);
            a[2]=(__bf16)(f0.z*ATTN_SCALE);  a[3]=(__bf16)(f0.w*ATTN_SCALE);
            a[4]=(__bf16)(f1.x*ATTN_SCALE);  a[5]=(__bf16)(f1.y*ATTN_SCALE);
            a[6]=(__bf16)(f1.z*ATTN_SCALE);  a[7]=(__bf16)(f1.w*ATTN_SCALE);
            a[8]=(__bf16)(f2.x*ATTN_SCALE);  a[9]=(__bf16)(f2.y*ATTN_SCALE);
            a[10]=(__bf16)(f2.z*ATTN_SCALE); a[11]=(__bf16)(f2.w*ATTN_SCALE);
            a[12]=(__bf16)(f3.x*ATTN_SCALE); a[13]=(__bf16)(f3.y*ATTN_SCALE);
            a[14]=(__bf16)(f3.z*ATTN_SCALE); a[15]=(__bf16)(f3.w*ATTN_SCALE);
            qa[c] = a;
        }
    }

    const v8f vzero = {0.f, 0.f, 0.f, 0.f, 0.f, 0.f, 0.f, 0.f};
    v8f acc[8];
    #pragma unroll
    for (int j = 0; j < 8; ++j) acc[j] = vzero;
    float rm[8], rl[8];
    #pragma unroll
    for (int r = 0; r < 8; ++r) { rm[r] = NEG_BIG; rl[r] = 0.0f; }

    const int nIter = (qbase + QTILE + KTILE - 1) / KTILE;   // causal bound

    // Prologue: stage block 0.
    load_tile(k, v, Kt[0], Vt[0], b, kvh, 0, tid);
    __syncthreads();

    v8f s0, s1;
    // Full (unmasked) blocks: every block except the diagonal one.
    for (int it = 0; it < nIter - 1; ++it) {
        const int kbase = it * KTILE;
        const int cur = it & 1;
        qk_block(Kt[cur], qa, s0, s1, nl, hs);
        // Prefetch next block while this wave's QK WMMAs drain.
        load_tile(k, v, Kt[cur ^ 1], Vt[cur ^ 1], b, kvh, kbase + KTILE, tid);
        softmax_pv_block<false>(s0, s1, Vt[cur], Pt[wave], acc, rm, rl,
                                kbase, qbase, nl, hs);
        // One barrier: publishes next buffer, retires the previous one.
        __syncthreads();
    }
    // Diagonal (masked) block.
    {
        const int it = nIter - 1;
        const int kbase = it * KTILE;
        const int cur = it & 1;
        qk_block(Kt[cur], qa, s0, s1, nl, hs);
        softmax_pv_block<true>(s0, s1, Vt[cur], Pt[wave], acc, rm, rl,
                               kbase, qbase, nl, hs);
    }

    // ---- epilogue: normalize and store (lanes 0..15 write 64B contiguous runs)
    #pragma unroll
    for (int r = 0; r < 8; ++r) {
        const float inv = 1.0f / rl[r];
        const int rowg = qbase + 8 * hs + r;
        float* orow = out + (((size_t)(b * SEQ + rowg)) * NH + head) * HD;
        #pragma unroll
        for (int j = 0; j < 8; ++j)
            orow[16 * j + nl] = acc[j][r] * inv;
    }
}

extern "C" void kernel_launch(void* const* d_in, const int* in_sizes, int n_in,
                              void* d_out, int out_size, void* d_ws, size_t ws_size,
                              hipStream_t stream) {
    (void)in_sizes; (void)n_in; (void)out_size; (void)d_ws; (void)ws_size;
    const float* q   = (const float*)d_in[0];
    const float* k   = (const float*)d_in[1];
    const float* v   = (const float*)d_in[2];
    const int* slot  = (const int*)d_in[5];

    float* out = (float*)d_out;
    float* kc  = out + (size_t)NTOK * NH * HD;
    float* vc  = kc + (size_t)NTOK * NKV * HD;

    // KV-cache scatter: slot_mapping covers every token, fully initializes kc/vc.
    const int total = NTOK * NKV * HD;            // 4,194,304
    kv_scatter_kernel<<<total / 256, 256, 0, stream>>>(k, v, slot, kc, vc, total);

    // Flash attention: one block per (batch, kv-head, 16-row q tile); 4 waves
    // = the 4 GQA q-heads sharing that kv head.
    const int nblk = (NTOK / SEQ) * NKV * (SEQ / QTILE);   // 2048
    fa_fwd_kernel<<<nblk, 128, 0, stream>>>(q, k, v, out);
}
